// NyAttenLayer_28922309771267
// MI455X (gfx1250) — compile-verified, matrix operand order checked by reading, not run on hardware
//
#include <hip/hip_runtime.h>

// ---------------------------------------------------------------------------
// Nystrom attention layer for MI455X (gfx1250, wave32, WMMA).
// Big GEMMs: bf16 WMMA (v_wmma_f32_16x16x32_bf16), fp32 accumulate,
//            double-buffered LDS staging, branch-free global->LDS copies.
// Moore-Penrose pinv Newton-Schulz: fp32 WMMA (v_wmma_f32_16x16x4_f32).
// ---------------------------------------------------------------------------

typedef __bf16 bf16;
typedef __attribute__((ext_vector_type(16))) __bf16 v16bf;
typedef __attribute__((ext_vector_type(8)))  float  v8f;
typedef __attribute__((ext_vector_type(2)))  float  v2f;

#define DEVI __device__ __forceinline__

DEVI float bf2f(bf16 x) { return (float)x; }
DEVI bf16  f2bf(float x) { return (bf16)x; }

// Problem constants (match reference).
static constexpr int NB    = 2;       // batch
static constexpr int NSEQ  = 8192;    // tokens
static constexpr int H     = 8;       // heads
static constexpr int DH    = 64;      // dim head
static constexpr int BH    = NB * H;  // 16 batched heads
static constexpr int LM    = 256;     // landmarks
static constexpr int DIM   = 512;
static constexpr int KER   = 33;

// ---------------------------------------------------------------------------
// Epilogue parameter bundle for the bf16 NT GEMM.
// ---------------------------------------------------------------------------
struct Epi {
    float*      Cf;       // EPI 0: f32 C
    bf16*       Cb;       // EPI 1: bf16 C
    long long   strideC;  // per-batch C stride (elements)
    int         ldc;
    bf16 *q, *k, *v, *vT; // EPI 2 (qkv scatter)
    const float *xres, *bias; // EPI 3 (final residual)
    float*      yout;
    float       scale;
};

// ---------------------------------------------------------------------------
// Generic NT GEMM:  C[M,N] = A[M,K] * Bt[N,K]^T   (bf16 in, f32 accum)
// Block: 128 threads = 4 waves; block tile 64x64; each wave 32x32 (2x2 WMMA).
// K staged through double-buffered LDS in chunks of 32.
// All dims multiples of 64/64/32 -> no edge guards, EXEC stays all-ones.
// ---------------------------------------------------------------------------
template <int EPI>
__global__ __launch_bounds__(128)
void gemm_bf16_nt(const bf16* __restrict__ A, const bf16* __restrict__ Bt,
                  int Mdim, int Ndim, int Kdim, int lda, int ldb,
                  long long sA, long long sB, Epi ep)
{
    __shared__ __align__(16) bf16 As[2][64][40]; // 40-halve stride: 16B aligned rows
    __shared__ __align__(16) bf16 Bs[2][64][40];

    const int tid  = threadIdx.x;
    const int lane = tid & 31;
    const int wave = tid >> 5;
    const int wr   = wave >> 1;     // wave row within block tile (0..1)
    const int wc   = wave & 1;      // wave col within block tile (0..1)
    const int rowBase = blockIdx.y * 64;
    const int colBase = blockIdx.x * 64;
    const int bz = blockIdx.z;

    A  += (long long)bz * sA;
    Bt += (long long)bz * sB;

    // Branch-free staging coordinates: each thread owns two 16B chunks per tile.
    const int row0 = tid >> 2;          // rows 0..31
    const int row1 = row0 + 32;         // rows 32..63
    const int ccs  = (tid & 3) << 3;    // halve offset within the 32-wide chunk
    const bf16* Ag0 = A  + (size_t)(rowBase + row0) * lda + ccs;
    const bf16* Ag1 = A  + (size_t)(rowBase + row1) * lda + ccs;
    const bf16* Bg0 = Bt + (size_t)(colBase + row0) * ldb + ccs;
    const bf16* Bg1 = Bt + (size_t)(colBase + row1) * ldb + ccs;

    const int r  = lane & 15;
    const int hf = lane >> 4;

    v8f acc[2][2] = {};

    // Prologue: stage K-chunk 0 into buffer 0.
    *(uint4*)&As[0][row0][ccs] = *(const uint4*)(Ag0);
    *(uint4*)&As[0][row1][ccs] = *(const uint4*)(Ag1);
    *(uint4*)&Bs[0][row0][ccs] = *(const uint4*)(Bg0);
    *(uint4*)&Bs[0][row1][ccs] = *(const uint4*)(Bg1);
    __syncthreads();

    int buf = 0;
    for (int k0 = 0; k0 < Kdim; k0 += 32) {
        // Prefetch next K-chunk into registers while WMMA consumes `buf`.
        const bool hasNext = (k0 + 32) < Kdim;   // wave-uniform -> scalar branch
        uint4 ra0, ra1, rb0, rb1;
        if (hasNext) {
            ra0 = *(const uint4*)(Ag0 + k0 + 32);
            ra1 = *(const uint4*)(Ag1 + k0 + 32);
            rb0 = *(const uint4*)(Bg0 + k0 + 32);
            rb1 = *(const uint4*)(Bg1 + k0 + 32);
        }

        // Per-lane WMMA fragments (ISA 16-bit 16x32 A layout; B mirrored).
        v16bf af[2], bfv[2];
        #pragma unroll
        for (int t = 0; t < 2; ++t) {
            const int ar = wr * 32 + t * 16 + r;
            const int br = wc * 32 + t * 16 + r;
            #pragma unroll
            for (int e = 0; e < 8; ++e) {
                af[t][e]      = As[buf][ar][hf * 8 + e];
                af[t][8 + e]  = As[buf][ar][16 + hf * 8 + e];
                bfv[t][e]     = Bs[buf][br][hf * 8 + e];
                bfv[t][8 + e] = Bs[buf][br][16 + hf * 8 + e];
            }
        }
        #pragma unroll
        for (int ti = 0; ti < 2; ++ti)
            #pragma unroll
            for (int tj = 0; tj < 2; ++tj)
                acc[ti][tj] = __builtin_amdgcn_wmma_f32_16x16x32_bf16(
                    false, af[ti], false, bfv[tj],
                    (short)0, acc[ti][tj], false, false);

        if (hasNext) {
            const int nb = buf ^ 1;
            *(uint4*)&As[nb][row0][ccs] = ra0;
            *(uint4*)&As[nb][row1][ccs] = ra1;
            *(uint4*)&Bs[nb][row0][ccs] = rb0;
            *(uint4*)&Bs[nb][row1][ccs] = rb1;
            __syncthreads();
            buf = nb;
        }
    }

    // Epilogue / store.  C 16x16 f32 layout: lane<16 -> rows 0..7, col=lane;
    // lane>=16 -> rows 8..15, col=lane-16.
    #pragma unroll
    for (int ti = 0; ti < 2; ++ti) {
        #pragma unroll
        for (int tj = 0; tj < 2; ++tj) {
            #pragma unroll
            for (int j = 0; j < 8; ++j) {
                const int rr = rowBase + wr * 32 + ti * 16 + (hf ? 8 + j : j);
                const int cc = colBase + wc * 32 + tj * 16 + r;
                const float val = acc[ti][tj][j];
                if constexpr (EPI == 0) {
                    ep.Cf[(long long)bz * ep.strideC + (size_t)rr * ep.ldc + cc] = val;
                } else if constexpr (EPI == 1) {
                    ep.Cb[(long long)bz * ep.strideC + (size_t)rr * ep.ldc + cc] = f2bf(val);
                } else if constexpr (EPI == 2) {
                    // QKV scatter: col -> (which, head, d); row -> (b, i)
                    const int which = cc >> 9;
                    const int h  = (cc >> 6) & 7;
                    const int d  = cc & 63;
                    const int bb = rr >> 13;
                    const int i  = rr & 8191;
                    const size_t base = ((size_t)(bb * 8 + h) * NSEQ + i) * DH + d;
                    if (which == 0) {
                        ep.q[base] = f2bf(val * ep.scale);
                    } else if (which == 1) {
                        ep.k[base] = f2bf(val);
                    } else {
                        ep.v[base] = f2bf(val);
                        ep.vT[((size_t)(bb * 8 + h) * DH + d) * NSEQ + i] = f2bf(val);
                    }
                } else { // EPI == 3 : y = x + o*Wout^T + bias
                    const size_t idx = (size_t)rr * DIM + cc;
                    ep.yout[idx] = ep.xres[idx] + ep.bias[cc] + val;
                }
            }
        }
    }
}

// ---------------------------------------------------------------------------
// fp32 WMMA GEMM:  C = dcoef*I + scoef*(A[M,K] * B[K,N])     (row-major both)
// Block (32,4): 4 waves, each one 16x16 tile; K loop step 4, unrolled x4.
// MODE 0: f32 store.   MODE 1: bf16 transposed store (CT[N][ldc]).
// ---------------------------------------------------------------------------
template <int MODE>
__global__ __launch_bounds__(128)
void gemm_f32_wmma(const float* __restrict__ A, const float* __restrict__ B,
                   int Kdim, int lda, int ldb,
                   long long sA, long long sB, long long sC,
                   float dcoef, float scoef,
                   float* __restrict__ Cf, bf16* __restrict__ CTb, int ldc)
{
    const int lane = threadIdx.x;
    const int wave = threadIdx.y;
    const int tileRow = blockIdx.y * 64 + wave * 16;
    const int tileCol = blockIdx.x * 16;
    const int bz = blockIdx.z;
    A += (long long)bz * sA;
    B += (long long)bz * sB;

    const int r  = lane & 15;
    const int hf = lane >> 4;

    const float* Arow = A + (size_t)(tileRow + r) * lda + hf * 2; // contiguous pair
    const float* Bcol = B + (size_t)(hf * 2) * ldb + tileCol + r; // ldb-strided pair

    v8f acc = {};
    #pragma unroll 4
    for (int k = 0; k < Kdim; k += 4) {
        v2f a, b;
        a.x = Arow[k];
        a.y = Arow[k + 1];
        b.x = Bcol[(size_t)k * ldb];
        b.y = Bcol[(size_t)k * ldb + ldb];
        acc = __builtin_amdgcn_wmma_f32_16x16x4_f32(
            false, a, false, b, (short)0, acc, false, false);
    }
    #pragma unroll
    for (int j = 0; j < 8; ++j) {
        const int rr = tileRow + (hf ? 8 + j : j);
        const int cc = tileCol + r;
        float val = scoef * acc[j];
        if constexpr (MODE == 0) {
            if (rr == cc) val += dcoef;
            Cf[(long long)bz * sC + (size_t)rr * ldc + cc] = val;
        } else {
            CTb[(long long)bz * sC + (size_t)cc * ldc + rr] = f2bf(val);
        }
    }
}

// ---------------------------------------------------------------------------
// Helpers
// ---------------------------------------------------------------------------
__global__ void f32_to_bf16(const float* __restrict__ s, bf16* __restrict__ d, size_t n)
{
    const size_t i = (size_t)blockIdx.x * 256 + threadIdx.x;
    if (i < n) d[i] = f2bf(s[i]);
}

// dst[N][K] = (bf16) src[K][N]
__global__ void transpose_to_bf16(const float* __restrict__ s, bf16* __restrict__ d,
                                  int K, int N)
{
    const size_t i = (size_t)blockIdx.x * 256 + threadIdx.x;
    if (i < (size_t)K * N) {
        const int n = (int)(i / K);
        const int k = (int)(i % K);
        d[i] = f2bf(s[(size_t)k * N + n]);
    }
}

// landmark means: dst[bh][j][d] = mean_{t<32} src[bh][j*32+t][d]
__global__ void landmark_mean(const bf16* __restrict__ src, bf16* __restrict__ dst)
{
    const size_t idx = (size_t)blockIdx.x * 256 + threadIdx.x; // BH*LM*DH
    const int d  = (int)(idx & 63);
    const int j  = (int)((idx >> 6) & 255);
    const int bh = (int)(idx >> 14);
    const bf16* sp = src + ((size_t)bh * NSEQ + (size_t)j * 32) * DH + d;
    float s = 0.f;
    #pragma unroll
    for (int t = 0; t < 32; ++t) s += bf2f(sp[(size_t)t * DH]);
    dst[idx] = f2bf(s * (1.f / 32.f));
}

__global__ __launch_bounds__(256)
void softmax_bf16(bf16* __restrict__ p, int len)
{
    __shared__ float red[256];
    bf16* rp = p + (size_t)blockIdx.x * len;
    const int tid = threadIdx.x;
    float mx = -3.0e38f;
    for (int j = tid; j < len; j += 256) mx = fmaxf(mx, bf2f(rp[j]));
    red[tid] = mx; __syncthreads();
    for (int s = 128; s > 0; s >>= 1) {
        if (tid < s) red[tid] = fmaxf(red[tid], red[tid + s]);
        __syncthreads();
    }
    mx = red[0]; __syncthreads();
    float sum = 0.f;
    for (int j = tid; j < len; j += 256) sum += __expf(bf2f(rp[j]) - mx);
    red[tid] = sum; __syncthreads();
    for (int s = 128; s > 0; s >>= 1) {
        if (tid < s) red[tid] += red[tid + s];
        __syncthreads();
    }
    const float inv = 1.f / red[0];
    for (int j = tid; j < len; j += 256)
        rp[j] = f2bf(__expf(bf2f(rp[j]) - mx) * inv);
}

__global__ __launch_bounds__(256)
void softmax_f32(float* __restrict__ p, int len)
{
    __shared__ float red[256];
    float* rp = p + (size_t)blockIdx.x * len;
    const int tid = threadIdx.x;
    float mx = -3.0e38f;
    for (int j = tid; j < len; j += 256) mx = fmaxf(mx, rp[j]);
    red[tid] = mx; __syncthreads();
    for (int s = 128; s > 0; s >>= 1) {
        if (tid < s) red[tid] = fmaxf(red[tid], red[tid + s]);
        __syncthreads();
    }
    mx = red[0]; __syncthreads();
    float sum = 0.f;
    for (int j = tid; j < len; j += 256) sum += __expf(rp[j] - mx);
    red[tid] = sum; __syncthreads();
    for (int s = 128; s > 0; s >>= 1) {
        if (tid < s) red[tid] += red[tid + s];
        __syncthreads();
    }
    const float inv = 1.f / red[0];
    for (int j = tid; j < len; j += 256)
        rp[j] = __expf(rp[j] - mx) * inv;
}

// Global max over (row-sums, col-sums) of |attn2|; values >=0 -> uint bit max.
__global__ __launch_bounds__(256)
void attn2_absmax(const float* __restrict__ a2, unsigned int* __restrict__ scal)
{
    const int bh = blockIdx.x;
    const int j  = threadIdx.x;
    const float* m = a2 + (size_t)bh * LM * LM;
    float rs = 0.f, cs = 0.f;
    for (int t = 0; t < LM; ++t) {
        rs += fabsf(m[(size_t)j * LM + t]);
        cs += fabsf(m[(size_t)t * LM + j]);
    }
    atomicMax(&scal[0], __float_as_uint(rs)); // max row-sum  ("col" in ref)
    atomicMax(&scal[1], __float_as_uint(cs)); // max col-sum  ("row" in ref)
}

// z = attn2^T / (col*row)
__global__ void zinit(const float* __restrict__ a2, float* __restrict__ z,
                      const unsigned int* __restrict__ scal)
{
    const size_t idx = (size_t)blockIdx.x * 256 + threadIdx.x; // BH*LM*LM
    const float denom = __uint_as_float(scal[0]) * __uint_as_float(scal[1]);
    const int bh = (int)(idx >> 16);
    const int rem = (int)(idx & 65535);
    const int i = rem >> 8, j = rem & 255;
    z[idx] = a2[(size_t)bh * LM * LM + (size_t)j * LM + i] / denom;
}

// dst = d*I + s*src   (batched 256x256)
__global__ void diag_axpy(const float* __restrict__ src, float* __restrict__ dst,
                          float d, float s)
{
    const size_t idx = (size_t)blockIdx.x * 256 + threadIdx.x;
    const int rem = (int)(idx & 65535);
    const int i = rem >> 8, j = rem & 255;
    dst[idx] = ((i == j) ? d : 0.f) + s * src[idx];
}

// depthwise conv over sequence + merge heads into [b][n][h*64+d] bf16
__global__ __launch_bounds__(256)
void conv_merge(const float* __restrict__ outh, const bf16* __restrict__ v,
                const float* __restrict__ wconv, bf16* __restrict__ obf)
{
    const size_t idx = (size_t)blockIdx.x * 256 + threadIdx.x; // BH*NSEQ*DH
    const int d  = (int)(idx & 63);
    const int i  = (int)((idx >> 6) & 8191);
    const int bh = (int)(idx >> 19);
    const int h  = bh & 7;
    const int bb = bh >> 3;
    float s = outh[idx];
    const bf16* vp = v + (size_t)bh * NSEQ * DH + d;
    #pragma unroll
    for (int kk = 0; kk < KER; ++kk) {
        const int t = i + kk - (KER / 2);
        if (t >= 0 && t < NSEQ)
            s += wconv[h * KER + kk] * bf2f(vp[(size_t)t * DH]);
    }
    obf[((size_t)bb * NSEQ + i) * DIM + h * DH + d] = f2bf(s);
}

// ---------------------------------------------------------------------------
// Host orchestration
// ---------------------------------------------------------------------------
extern "C" void kernel_launch(void* const* d_in, const int* in_sizes, int n_in,
                              void* d_out, int out_size, void* d_ws, size_t ws_size,
                              hipStream_t stream)
{
    const float* x      = (const float*)d_in[0];
    const float* w_qkv  = (const float*)d_in[1];
    const float* w_out  = (const float*)d_in[2];
    const float* b_out  = (const float*)d_in[3];
    const float* w_conv = (const float*)d_in[4];
    float* yout = (float*)d_out;

    char* base = (char*)d_ws;
    size_t off = 0;
    auto alloc = [&](size_t bytes) -> char* {
        char* p = base + off;
        off = (off + bytes + 255) & ~(size_t)255;
        return p;
    };

    const size_t tokD = (size_t)NB * NSEQ * DIM;       // 16384*512
    bf16* xbf   = (bf16*)alloc(tokD * 2);
    bf16* wqkvT = (bf16*)alloc((size_t)3 * DIM * DIM * 2);
    bf16* woutT = (bf16*)alloc((size_t)DIM * DIM * 2);
    bf16* qb    = (bf16*)alloc((size_t)BH * NSEQ * DH * 2);
    bf16* kb    = (bf16*)alloc((size_t)BH * NSEQ * DH * 2);
    bf16* vb    = (bf16*)alloc((size_t)BH * NSEQ * DH * 2);
    bf16* vT    = (bf16*)alloc((size_t)BH * DH * NSEQ * 2);
    bf16* ql    = (bf16*)alloc((size_t)BH * LM * DH * 2);
    bf16* kl    = (bf16*)alloc((size_t)BH * LM * DH * 2);
    bf16* attn1 = (bf16*)alloc((size_t)BH * NSEQ * LM * 2);
    bf16* attn3 = (bf16*)alloc((size_t)BH * LM * NSEQ * 2);
    float* attn2 = (float*)alloc((size_t)BH * LM * LM * 4);
    float* zA   = (float*)alloc((size_t)BH * LM * LM * 4);
    float* zB   = (float*)alloc((size_t)BH * LM * LM * 4);
    float* xz   = (float*)alloc((size_t)BH * LM * LM * 4);
    float* ta   = (float*)alloc((size_t)BH * LM * LM * 4);
    float* tb   = (float*)alloc((size_t)BH * LM * LM * 4);
    float* mid1 = (float*)alloc((size_t)BH * LM * DH * 4);
    bf16* mid2T = (bf16*)alloc((size_t)BH * DH * LM * 2);
    unsigned int* scal = (unsigned int*)alloc(64);
    // Aliases: outh (f32, 33.5MB) reuses attn3's region (67MB, dead by then);
    // obf (bf16 re-layout) reuses xbf (dead after QKV GEMM).
    float* outh = (float*)attn3;
    bf16*  obf  = xbf;

    hipMemsetAsync(scal, 0, 8, stream);

    // ---- convert inputs -------------------------------------------------
    f32_to_bf16<<<(unsigned)((tokD + 255) / 256), 256, 0, stream>>>(x, xbf, tokD);
    transpose_to_bf16<<<(3 * DIM * DIM + 255) / 256, 256, 0, stream>>>(
        w_qkv, wqkvT, DIM, 3 * DIM);
    transpose_to_bf16<<<(DIM * DIM + 255) / 256, 256, 0, stream>>>(
        w_out, woutT, DIM, DIM);

    // ---- QKV projection (fused scale + head scatter + V^T copy) ---------
    {
        Epi e{}; e.q = qb; e.k = kb; e.v = vb; e.vT = vT; e.scale = 0.125f; // 64^-0.5
        gemm_bf16_nt<2><<<dim3((3 * DIM) / 64, (NB * NSEQ) / 64, 1), 128, 0, stream>>>(
            xbf, wqkvT, NB * NSEQ, 3 * DIM, DIM, DIM, DIM, 0, 0, e);
    }

    // ---- landmark means -------------------------------------------------
    landmark_mean<<<(BH * LM * DH) / 256, 256, 0, stream>>>(qb, ql);
    landmark_mean<<<(BH * LM * DH) / 256, 256, 0, stream>>>(kb, kl);

    // ---- sim1 = q @ kl^T -> softmax -> attn1 (bf16) ---------------------
    {
        Epi e{}; e.Cb = attn1; e.ldc = LM; e.strideC = (long long)NSEQ * LM;
        gemm_bf16_nt<1><<<dim3(LM / 64, NSEQ / 64, BH), 128, 0, stream>>>(
            qb, kl, NSEQ, LM, DH, DH, DH,
            (long long)NSEQ * DH, (long long)LM * DH, e);
    }
    softmax_bf16<<<BH * NSEQ, 256, 0, stream>>>(attn1, LM);

    // ---- sim2 = ql @ kl^T -> softmax -> attn2 (f32) ---------------------
    {
        Epi e{}; e.Cf = attn2; e.ldc = LM; e.strideC = (long long)LM * LM;
        gemm_bf16_nt<0><<<dim3(LM / 64, LM / 64, BH), 128, 0, stream>>>(
            ql, kl, LM, LM, DH, DH, DH,
            (long long)LM * DH, (long long)LM * DH, e);
    }
    softmax_f32<<<BH * LM, 256, 0, stream>>>(attn2, LM);

    // ---- sim3 = ql @ k^T -> softmax -> attn3 (bf16) ---------------------
    {
        Epi e{}; e.Cb = attn3; e.ldc = NSEQ; e.strideC = (long long)LM * NSEQ;
        gemm_bf16_nt<1><<<dim3(NSEQ / 64, LM / 64, BH), 128, 0, stream>>>(
            ql, kb, LM, NSEQ, DH, DH, DH,
            (long long)LM * DH, (long long)NSEQ * DH, e);
    }
    softmax_bf16<<<BH * LM, 256, 0, stream>>>(attn3, NSEQ);

    // ---- Moore-Penrose pinv of attn2 (fp32 WMMA Newton-Schulz) ----------
    attn2_absmax<<<BH, 256, 0, stream>>>(attn2, scal);
    zinit<<<(BH * LM * LM) / 256, 256, 0, stream>>>(attn2, zA, scal);

    const long long sM = (long long)LM * LM;
    float* zc = zA; float* zn = zB;
    for (int it = 0; it < 6; ++it) {
        // xz = attn2 @ z
        gemm_f32_wmma<0><<<dim3(LM / 16, LM / 64, BH), dim3(32, 4), 0, stream>>>(
            attn2, zc, LM, LM, LM, sM, sM, sM, 0.f, 1.f, xz, nullptr, LM);
        // ta = 7I - xz
        diag_axpy<<<(BH * LM * LM) / 256, 256, 0, stream>>>(xz, ta, 7.f, -1.f);
        // tb = 15I - xz@ta
        gemm_f32_wmma<0><<<dim3(LM / 16, LM / 64, BH), dim3(32, 4), 0, stream>>>(
            xz, ta, LM, LM, LM, sM, sM, sM, 15.f, -1.f, tb, nullptr, LM);
        // ta = 13I - xz@tb
        gemm_f32_wmma<0><<<dim3(LM / 16, LM / 64, BH), dim3(32, 4), 0, stream>>>(
            xz, tb, LM, LM, LM, sM, sM, sM, 13.f, -1.f, ta, nullptr, LM);
        // zn = 0.25 * z@ta
        gemm_f32_wmma<0><<<dim3(LM / 16, LM / 64, BH), dim3(32, 4), 0, stream>>>(
            zc, ta, LM, LM, LM, sM, sM, sM, 0.f, 0.25f, zn, nullptr, LM);
        float* t = zc; zc = zn; zn = t;
    }

    // ---- mid1 = attn3 @ v  (NT with vT), f32 ----------------------------
    {
        Epi e{}; e.Cf = mid1; e.ldc = DH; e.strideC = (long long)LM * DH;
        gemm_bf16_nt<0><<<dim3(DH / 64, LM / 64, BH), 128, 0, stream>>>(
            attn3, vT, LM, DH, NSEQ, NSEQ, NSEQ,
            (long long)LM * NSEQ, (long long)DH * NSEQ, e);
    }

    // ---- mid2T = (z_final @ mid1)^T, bf16 [64][256] ----------------------
    gemm_f32_wmma<1><<<dim3(DH / 16, LM / 64, BH), dim3(32, 4), 0, stream>>>(
        zc, mid1, LM, LM, DH, sM, (long long)LM * DH, (long long)DH * LM,
        0.f, 1.f, nullptr, mid2T, LM);

    // ---- outh = attn1 @ mid2  (NT with mid2T), f32 ----------------------
    {
        Epi e{}; e.Cf = outh; e.ldc = DH; e.strideC = (long long)NSEQ * DH;
        gemm_bf16_nt<0><<<dim3(DH / 64, NSEQ / 64, BH), 128, 0, stream>>>(
            attn1, mid2T, NSEQ, DH, LM, LM, LM,
            (long long)NSEQ * LM, (long long)DH * LM, e);
    }

    // ---- depthwise conv residual + merge heads -> obf (bf16) ------------
    conv_merge<<<(BH * NSEQ * DH) / 256, 256, 0, stream>>>(outh, vb, w_conv, obf);

    // ---- final projection + bias + input residual -----------------------
    {
        Epi e{}; e.xres = x; e.bias = b_out; e.yout = yout;
        gemm_bf16_nt<3><<<dim3(DIM / 64, (NB * NSEQ) / 64, 1), 128, 0, stream>>>(
            obf, woutT, NB * NSEQ, DIM, DIM, DIM, DIM, 0, 0, e);
    }
}
